// GATConv_88072599372363
// MI455X (gfx1250) — matrix operand order repaired and checked by reading, not compiled, not run on hardware
//
#include <hip/hip_runtime.h>
#include <hip/hip_bf16.h>

#define N_NODES   50000
#define DEG       16
#define IN_FEATS  256
#define NUM_HEADS 4
#define OUT_FEATS 64
#define OUT_COLS  (NUM_HEADS * OUT_FEATS)   // 256
#define NEG_SLOPE 0.2f
#define NODES_PER_BLK 4

typedef __attribute__((ext_vector_type(16))) _Float16 v16h;
typedef __attribute__((ext_vector_type(8)))  _Float16 v8h;
typedef __attribute__((ext_vector_type(8)))  float    v8f;

// ---------------- workspace layout ----------------
// [0, H_BYTES)           : h  (N x 256 f32)  = 51,200,000 B  (L2-resident)
// [H_BYTES, +WT_BYTES)   : Wt (256 x 256 f16, transposed)
// [.., +AR_BYTES)        : attn_row (N x 4 f32)
// [.., +AR_BYTES)        : attn_col (N x 4 f32)
static constexpr size_t H_BYTES  = (size_t)N_NODES * OUT_COLS * sizeof(float);
static constexpr size_t WT_BYTES = (size_t)IN_FEATS * OUT_COLS * sizeof(_Float16);
static constexpr size_t AR_BYTES = (size_t)N_NODES * NUM_HEADS * sizeof(float);

// ---------------------------------------------------------------------------
// Kernel 1: Wt[c*256 + k] = (f16) W[k*256 + c]   (transpose + downconvert)
// ---------------------------------------------------------------------------
__global__ void k_prep_w(const float* __restrict__ W, _Float16* __restrict__ Wt) {
    int i = blockIdx.x * blockDim.x + threadIdx.x;   // 65536 total
    int c = i >> 8;
    int k = i & 255;
    Wt[i] = (_Float16)W[k * OUT_COLS + c];
}

// ---------------------------------------------------------------------------
// Kernel 2: h = feat @ W via v_wmma_f32_16x16x32_f16.
// Block = 256 threads = 8 waves. Block owns a 16-node tile; the fp32 feat tile
// is converted to f16 in LDS ONCE per block (instead of once per wave), then
// each wave feeds its A fragments from LDS (ds_load_b128) while B fragments
// stream from the tiny L0/L2-resident Wt. Wave w owns output columns
// [w*32, w*32+32) = two 16x16 tiles; K loop: 8 steps of 32.
// ---------------------------------------------------------------------------
#define APAD 8   // 8 halfs = 16B row padding -> 528B row stride, rotates banks
__global__ void k_gemm(const float* __restrict__ feat,
                       const _Float16* __restrict__ Wt,
                       float* __restrict__ h) {
    __shared__ _Float16 sA[16][IN_FEATS + APAD];   // ~8.25 KB of 320 KB

    const int tileM = blockIdx.x * 16;

    // ---- cooperative load + f32->f16 convert of the 16x256 feat tile ----
    {
        const int r  = threadIdx.x >> 4;          // 0..15
        const int c0 = (threadIdx.x & 15) * 16;   // 16 halfs per thread
        const float4* src = (const float4*)(feat + (size_t)(tileM + r) * IN_FEATS + c0);
        float4 f0 = src[0], f1 = src[1], f2 = src[2], f3 = src[3];
        _Float16* dst = &sA[r][c0];
        dst[0]  = (_Float16)f0.x; dst[1]  = (_Float16)f0.y;
        dst[2]  = (_Float16)f0.z; dst[3]  = (_Float16)f0.w;
        dst[4]  = (_Float16)f1.x; dst[5]  = (_Float16)f1.y;
        dst[6]  = (_Float16)f1.z; dst[7]  = (_Float16)f1.w;
        dst[8]  = (_Float16)f2.x; dst[9]  = (_Float16)f2.y;
        dst[10] = (_Float16)f2.z; dst[11] = (_Float16)f2.w;
        dst[12] = (_Float16)f3.x; dst[13] = (_Float16)f3.y;
        dst[14] = (_Float16)f3.z; dst[15] = (_Float16)f3.w;
    }
    __syncthreads();

    const int wave = threadIdx.x >> 5;       // 0..7
    const int lane = threadIdx.x & 31;
    const int row  = lane & 15;              // A row / D column index
    const int hi   = lane >> 4;              // half-wave selector
    const int col0 = wave * 32;

    // B fragment base: lane's column = col0 + row; element e <-> K = k0 + hi*16 + e
    const _Float16* b0p = Wt + (size_t)(col0 + row) * IN_FEATS + hi * 16;
    const _Float16* b1p = b0p + (size_t)16 * IN_FEATS;

    v8f c0acc = {};
    v8f c1acc = {};

#pragma unroll
    for (int k0 = 0; k0 < IN_FEATS; k0 += 32) {
        // ---- A fragment (contiguous K order, matching B) from LDS:
        //   elements 0..7  <-> K = k0 + hi*16 + j      (one ds_load_b128)
        //   elements 8..15 <-> K = k0 + hi*16 + 8 + j  (one ds_load_b128)
        v8h alo = *(const v8h*)&sA[row][k0 + hi * 16];
        v8h ahi = *(const v8h*)&sA[row][k0 + hi * 16 + 8];
        v16h a  = __builtin_shufflevector(alo, ahi, 0, 1, 2, 3, 4, 5, 6, 7,
                                                    8, 9, 10, 11, 12, 13, 14, 15);

        v16h b0 = *(const v16h*)(b0p + k0);   // 32B aligned contiguous load
        v16h b1 = *(const v16h*)(b1p + k0);

        c0acc = __builtin_amdgcn_wmma_f32_16x16x32_f16(false, a, false, b0,
                                                       (short)0, c0acc, false, false);
        c1acc = __builtin_amdgcn_wmma_f32_16x16x32_f16(false, a, false, b1,
                                                       (short)0, c1acc, false, false);
    }

    // ---- D store (32-bit C/D 16x16): element r -> M = r + 8*hi, N = lane&15
    float* hbase = h + (size_t)(tileM + hi * 8) * OUT_COLS + col0 + row;
#pragma unroll
    for (int r = 0; r < 8; ++r) {
        hbase[(size_t)r * OUT_COLS]      = c0acc[r];
        hbase[(size_t)r * OUT_COLS + 16] = c1acc[r];
    }
}

// ---------------------------------------------------------------------------
// Kernel 3: per-(node, head) attention scores
// ---------------------------------------------------------------------------
__global__ void k_attn(const float* __restrict__ h,
                       const float* __restrict__ attn_l,
                       const float* __restrict__ attn_r,
                       float* __restrict__ arow,
                       float* __restrict__ acol) {
    int i = blockIdx.x * blockDim.x + threadIdx.x;
    if (i >= N_NODES * NUM_HEADS) return;
    int n  = i >> 2;
    int hd = i & 3;
    const float* hp = h + (size_t)n * OUT_COLS + hd * OUT_FEATS;
    const float* al = attn_l + hd * OUT_FEATS;
    const float* ar = attn_r + hd * OUT_FEATS;
    float sl = 0.f, sr = 0.f;
#pragma unroll
    for (int j = 0; j < OUT_FEATS; j += 4) {
        float4 hv = *(const float4*)(hp + j);
        float4 lv = *(const float4*)(al + j);
        float4 rv = *(const float4*)(ar + j);
        sl += hv.x * lv.x + hv.y * lv.y + hv.z * lv.z + hv.w * lv.w;
        sr += hv.x * rv.x + hv.y * rv.y + hv.z * rv.z + hv.w * rv.w;
    }
    arow[i] = sl;
    acol[i] = sr;
}

// ---------------------------------------------------------------------------
// Kernel 4: per-destination softmax + weighted gather aggregation.
// 4 nodes per 256-thread block; 64 threads x float4 per node so the gather
// uses global_load_b128 (4x fewer load issues than b32 for the ~860 MB of
// L2-resident gather traffic). 256 threads = 4 nodes x 4 heads x 16 edges
// maps one LeakyReLU score per thread; 16 threads run the 16-wide softmaxes.
// ---------------------------------------------------------------------------
__global__ void k_aggregate(const int* __restrict__ row_ptr,
                            const int* __restrict__ col_ind,
                            const float* __restrict__ h,
                            const float* __restrict__ arow,
                            const float* __restrict__ acol,
                            float* __restrict__ out) {
    const int t        = threadIdx.x;                 // 256
    const int nodeBase = blockIdx.x * NODES_PER_BLK;

    __shared__ int   s_src[NODES_PER_BLK][DEG];
    __shared__ float s_sc[NODES_PER_BLK][NUM_HEADS][DEG];
    __shared__ float s_alpha[NODES_PER_BLK][NUM_HEADS][DEG];

    if (t < NODES_PER_BLK * DEG) {
        int q = t >> 4, e = t & 15;
        s_src[q][e] = col_ind[row_ptr[nodeBase + q] + e];
    }
    __syncthreads();

    const int q  = t >> 6;            // local node for the gather phase
    const int c4 = (t & 63) * 4;      // starting feature column (float4)

    // Warm L0/L2 for the gathered rows while the softmax runs
    // (emits global_prefetch_b8 per lane).
#pragma unroll
    for (int e = 0; e < DEG; ++e)
        __builtin_prefetch(h + (size_t)s_src[q][e] * OUT_COLS + c4, 0, 1);

    {   // one LeakyReLU edge score per thread: (node, head, edge)
        int sq = t >> 6, hd = (t >> 4) & 3, e = t & 15;
        float sc = arow[(nodeBase + sq) * NUM_HEADS + hd]
                 + acol[s_src[sq][e] * NUM_HEADS + hd];
        s_sc[sq][hd][e] = (sc > 0.f) ? sc : NEG_SLOPE * sc;
    }
    __syncthreads();

    if (t < NODES_PER_BLK * NUM_HEADS) {   // 16 softmaxes of width 16
        int sq = t >> 2, hd = t & 3;
        float m = -INFINITY;
#pragma unroll
        for (int e = 0; e < DEG; ++e) m = fmaxf(m, s_sc[sq][hd][e]);
        float ex[DEG];
        float sum = 0.f;
#pragma unroll
        for (int e = 0; e < DEG; ++e) { ex[e] = __expf(s_sc[sq][hd][e] - m); sum += ex[e]; }
        float inv = 1.f / sum;
#pragma unroll
        for (int e = 0; e < DEG; ++e) s_alpha[sq][hd][e] = ex[e] * inv;
    }
    __syncthreads();

    const int hd = (t & 63) >> 4;
    float4 acc = make_float4(0.f, 0.f, 0.f, 0.f);
#pragma unroll
    for (int e = 0; e < DEG; ++e) {
        float  al = s_alpha[q][hd][e];
        float4 hv = *(const float4*)(h + (size_t)s_src[q][e] * OUT_COLS + c4);
        acc.x += al * hv.x;
        acc.y += al * hv.y;
        acc.z += al * hv.z;
        acc.w += al * hv.w;
    }
    *(float4*)(out + (size_t)(nodeBase + q) * OUT_COLS + c4) = acc;
}

// ---------------------------------------------------------------------------
extern "C" void kernel_launch(void* const* d_in, const int* in_sizes, int n_in,
                              void* d_out, int out_size, void* d_ws, size_t ws_size,
                              hipStream_t stream) {
    (void)in_sizes; (void)n_in; (void)out_size; (void)ws_size;

    const int*   row_ptr = (const int*)d_in[0];
    const int*   col_ind = (const int*)d_in[1];
    // d_in[2] (col_ptr) and d_in[3] (row_ind) are CSC mirrors, unused in fwd.
    const float* feat    = (const float*)d_in[4];
    const float* W       = (const float*)d_in[5];
    const float* attn_l  = (const float*)d_in[6];
    const float* attn_r  = (const float*)d_in[7];
    float*       out     = (float*)d_out;

    char*     ws   = (char*)d_ws;
    float*    h    = (float*)ws;
    _Float16* Wt   = (_Float16*)(ws + H_BYTES);
    float*    arow = (float*)(ws + H_BYTES + WT_BYTES);
    float*    acol = (float*)(ws + H_BYTES + WT_BYTES + AR_BYTES);

    k_prep_w<<<(IN_FEATS * OUT_COLS) / 256, 256, 0, stream>>>(W, Wt);
    k_gemm<<<N_NODES / 16, 256, 0, stream>>>(feat, Wt, h);
    k_attn<<<(N_NODES * NUM_HEADS + 255) / 256, 256, 0, stream>>>(h, attn_l, attn_r,
                                                                  arow, acol);
    k_aggregate<<<N_NODES / NODES_PER_BLK, 256, 0, stream>>>(row_ptr, col_ind, h,
                                                             arow, acol, out);
}
